// GNN_65953517797797
// MI455X (gfx1250) — matrix operand order repaired and checked by reading, not compiled, not run on hardware
//
#include <hip/hip_runtime.h>
#include <math.h>

// ---------- types ----------
typedef __bf16 bf16_t;
typedef __attribute__((ext_vector_type(16))) __bf16   v16bf;
typedef __attribute__((ext_vector_type(8)))  float    v8f;
typedef __attribute__((ext_vector_type(4)))  unsigned u32x4;

union FragU { u32x4 u[2]; v16bf v; };   // 32B fragment (8 VGPRs of bf16 pairs)

#define DH 128
#define TILE_ROWS 128
#define CHUNKS 16          // 16-byte chunks per 128-bf16 row

// XOR swizzle on 16B chunks: lanes reading 16 consecutive rows at a fixed
// chunk hit 16 distinct chunk slots -> conflict-free ds_load_b128.
__device__ __forceinline__ int swz(int row, int chunk) {
    return row * CHUNKS + (chunk ^ (row & 15));
}

// ---------- GEMM: C[nRows,128] = A[nRows,128](bf16) @ W[128,128](bf16) ----------
__global__ __launch_bounds__(256)
void k_gemm_bf16_wmma(const bf16_t* __restrict__ A, const bf16_t* __restrict__ W,
                      float* __restrict__ C, int nRows) {
    __shared__ u32x4 sA[TILE_ROWS * CHUNKS];   // 32 KB
    __shared__ u32x4 sB[DH       * CHUNKS];    // 32 KB
    const int tid = threadIdx.x;
    const int rowBase = blockIdx.x * TILE_ROWS;
    const u32x4* A4 = (const u32x4*)A;
    const u32x4* B4 = (const u32x4*)W;

    // cooperative stage via async copy unit: global -> LDS directly (ASYNCcnt),
    // swizzled LDS destination per lane. OOB rows (last block) zero-filled via DS.
    for (int q = tid; q < TILE_ROWS * CHUNKS; q += 256) {
        const int r = q >> 4, c = q & 15;
        const int gr = rowBase + r;
        const unsigned aOff = (unsigned)(size_t)(const void*)&sA[swz(r, c)];
        const unsigned bOff = (unsigned)(size_t)(const void*)&sB[swz(r, c)];
        const unsigned long long ga =
            (unsigned long long)(size_t)(A4 + (size_t)gr * CHUNKS + c);
        const unsigned long long gb = (unsigned long long)(size_t)(B4 + q);
        if (gr < nRows) {
            asm volatile("global_load_async_to_lds_b128 %0, %1, off"
                         :: "v"(aOff), "v"(ga) : "memory");
        } else {
            const u32x4 z = {0u, 0u, 0u, 0u};
            sA[swz(r, c)] = z;
        }
        asm volatile("global_load_async_to_lds_b128 %0, %1, off"
                     :: "v"(bOff), "v"(gb) : "memory");
    }
    asm volatile("s_wait_asynccnt 0x0" ::: "memory");
    __syncthreads();

    const int lane = tid & 31;
    const int wave = tid >> 5;          // 8 waves -> 8 row-tiles of 16
    const int half = lane >> 4;
    const int m    = lane & 15;
    const int arow = wave * 16 + m;     // A: lane holds row M=m

    v8f acc[8] = {};
    #pragma unroll
    for (int ks = 0; ks < 4; ++ks) {    // K = 4 * 32
        // A 16-bit 16x32 layout: v0-3 hold K = 8*half..+7, v4-7 hold K = 16+8*half..+7
        const int cA = ks * 4 + half;
        FragU fa;
        fa.u[0] = sA[swz(arow, cA)];
        fa.u[1] = sA[swz(arow, cA + 2)];
        // B layout: lane holds column N = nt*16 + m, element s holds K in the same order as A
        const bf16_t* sBh = (const bf16_t*)sB;
        #pragma unroll
        for (int nt = 0; nt < 8; ++nt) {
            FragU fb;
            const int bn = nt * 16 + m;
            #pragma unroll
            for (int s = 0; s < 16; ++s) {
                const int kk = ks * 32 + (s < 8 ? 8 * half + s : 8 + 8 * half + s);
                fb.v[s] = sBh[swz(kk, bn >> 3) * 8 + (bn & 7)];
            }
            acc[nt] = __builtin_amdgcn_wmma_f32_16x16x32_bf16(
                false, fa.v, false, fb.v, (short)0, acc[nt], false, false);
        }
    }

    // C/D layout: VGPR v, lane -> row M = v + 8*half, col N = m
    const int growBase = rowBase + wave * 16 + half * 8;
    #pragma unroll
    for (int nt = 0; nt < 8; ++nt) {
        #pragma unroll
        for (int v = 0; v < 8; ++v) {
            int rr = growBase + v;
            if (rr < nRows) C[(size_t)rr * DH + nt * 16 + m] = acc[nt][v];
        }
    }
}

// ---------- elementwise / graph kernels ----------
__global__ void k_cast_bf16(const float* __restrict__ src, bf16_t* __restrict__ dst, int n) {
    int t = blockIdx.x * 256 + threadIdx.x;
    if (t < n) dst[t] = (bf16_t)src[t];
}

__global__ void k_deg_init(float* __restrict__ deg, int n) {
    int t = blockIdx.x * 256 + threadIdx.x;
    if (t < n) deg[t] = 1.0f;                  // self-loop
}

__global__ void k_deg_count(const int* __restrict__ ei, float* __restrict__ deg, int E) {
    int t = blockIdx.x * 256 + threadIdx.x;
    if (t < E) unsafeAtomicAdd(&deg[ei[E + t]], 1.0f);   // dst row of edge_index
}

__global__ void k_dis(const float* __restrict__ deg, float* __restrict__ dis, int n) {
    int t = blockIdx.x * 256 + threadIdx.x;
    if (t < n) dis[t] = rsqrtf(deg[t]);
}

// agg = h * (dis^2) + bias   (self-loop term + bias folded in before scatter)
__global__ void k_agg_init(const float* __restrict__ h, const float* __restrict__ dis,
                           const float* __restrict__ bias, float* __restrict__ agg, int nf) {
    int t = blockIdx.x * 256 + threadIdx.x;
    if (t < nf) {
        int r = t >> 7;
        float d = dis[r];
        agg[t] = h[t] * d * d + bias[t & 127];
    }
}

// one edge per wave32; 4 floats per lane; f32 hardware atomics into agg[dst]
__global__ __launch_bounds__(256)
void k_scatter(const float* __restrict__ h, const float* __restrict__ dis,
               const int* __restrict__ ei, float* __restrict__ agg, int E) {
    const int wave = threadIdx.x >> 5;
    const int lane = threadIdx.x & 31;
    const int e = blockIdx.x * 8 + wave;
    if (e >= E) return;
    const int src = ei[e];
    const int dst = ei[E + e];
    const float coef = dis[src] * dis[dst];
    const float4 v = ((const float4*)(h + (size_t)src * DH))[lane];
    float* ad = agg + (size_t)dst * DH + lane * 4;
    unsafeAtomicAdd(ad + 0, v.x * coef);
    unsafeAtomicAdd(ad + 1, v.y * coef);
    unsafeAtomicAdd(ad + 2, v.z * coef);
    unsafeAtomicAdd(ad + 3, v.w * coef);
}

// relu + cast to bf16 (feeds next GEMM / pooling)
__global__ void k_relu_bf16(const float* __restrict__ agg, bf16_t* __restrict__ hb, int nf) {
    int t = blockIdx.x * 256 + threadIdx.x;
    if (t < nf) {
        float v = agg[t];
        hb[t] = (bf16_t)(v > 0.0f ? v : 0.0f);
    }
}

__global__ void k_pool_init(float* __restrict__ pooled, float* __restrict__ cnt, int G) {
    int t = blockIdx.x * 256 + threadIdx.x;
    if (t < G * DH) pooled[t] = 0.0f;
    if (t < G) cnt[t] = 0.0f;
}

__global__ void k_pool(const bf16_t* __restrict__ hb, const int* __restrict__ batch,
                       float* __restrict__ pooled, float* __restrict__ cnt, int nf) {
    int t = blockIdx.x * 256 + threadIdx.x;
    if (t >= nf) return;
    int node = t >> 7, f = t & 127;
    int g = batch[node];
    unsafeAtomicAdd(&pooled[g * DH + f], (float)hb[t]);
    if (f == 0) unsafeAtomicAdd(&cnt[g], 1.0f);
}

__global__ void k_head(const float* __restrict__ pooled, const float* __restrict__ cnt,
                       const float* __restrict__ Wfc, const float* __restrict__ bfc,
                       float* __restrict__ out, int G) {
    int t = blockIdx.x * 256 + threadIdx.x;
    if (t >= G * 8) return;
    int g = t >> 3, o = t & 7;
    float inv = 1.0f / fmaxf(cnt[g], 1.0f);
    float s = bfc[o];
    for (int k = 0; k < DH; ++k)
        s += pooled[g * DH + k] * inv * Wfc[k * 8 + o];
    out[t] = 1.0f / (1.0f + expf(-s));
}

// ---------- launcher ----------
extern "C" void kernel_launch(void* const* d_in, const int* in_sizes, int n_in,
                              void* d_out, int out_size, void* d_ws, size_t ws_size,
                              hipStream_t stream) {
    const float* x   = (const float*)d_in[0];
    const int*   ei  = (const int*)  d_in[1];   // [2,E] flat: [0..E)=src, [E..2E)=dst
    const int*   bat = (const int*)  d_in[2];
    const float* W1  = (const float*)d_in[3];
    const float* b1  = (const float*)d_in[4];
    const float* W2  = (const float*)d_in[5];
    const float* b2  = (const float*)d_in[6];
    const float* Wfc = (const float*)d_in[7];
    const float* bfc = (const float*)d_in[8];
    float* out = (float*)d_out;

    const int N = in_sizes[0] / DH;
    const int E = in_sizes[1] / 2;
    const int G = out_size / 8;
    const int nf = N * DH;

    // workspace carve-out (all buffers fully rewritten each call)
    char* w = (char*)d_ws;
    auto carve = [&](size_t bytes) {
        char* p = w;
        w += (bytes + 255) & ~(size_t)255;
        return p;
    };
    bf16_t* xbf = (bf16_t*)carve((size_t)nf * 2);
    bf16_t* hbf = (bf16_t*)carve((size_t)nf * 2);
    bf16_t* wt1 = (bf16_t*)carve((size_t)DH * DH * 2);
    bf16_t* wt2 = (bf16_t*)carve((size_t)DH * DH * 2);
    float*  h   = (float*) carve((size_t)nf * 4);
    float*  agg = (float*) carve((size_t)nf * 4);
    float*  deg = (float*) carve((size_t)N * 4);
    float*  dis = (float*) carve((size_t)N * 4);
    float*  pooled = (float*)carve((size_t)G * DH * 4);
    float*  cnt = (float*) carve((size_t)G * 4);
    (void)ws_size; (void)n_in;

    const dim3 B(256);
    const int gW  = (DH * DH + 255) / 256;
    const int gNF = (nf + 255) / 256;
    const int gN  = (N + 255) / 256;
    const int gE  = (E + 255) / 256;
    const int gEdgeWave = (E + 7) / 8;
    const int gGemm = (N + TILE_ROWS - 1) / TILE_ROWS;

    // precision prep
    k_cast_bf16<<<gW, B, 0, stream>>>(W1, wt1, DH * DH);
    k_cast_bf16<<<gW, B, 0, stream>>>(W2, wt2, DH * DH);
    k_cast_bf16<<<gNF, B, 0, stream>>>(x, xbf, nf);

    // gcn_norm
    k_deg_init<<<gN, B, 0, stream>>>(deg, N);
    k_deg_count<<<gE, B, 0, stream>>>(ei, deg, E);
    k_dis<<<gN, B, 0, stream>>>(deg, dis, N);

    // layer 1
    k_gemm_bf16_wmma<<<gGemm, B, 0, stream>>>(xbf, wt1, h, N);
    k_agg_init<<<gNF, B, 0, stream>>>(h, dis, b1, agg, nf);
    k_scatter<<<gEdgeWave, B, 0, stream>>>(h, dis, ei, agg, E);
    k_relu_bf16<<<gNF, B, 0, stream>>>(agg, hbf, nf);

    // layer 2
    k_gemm_bf16_wmma<<<gGemm, B, 0, stream>>>(hbf, wt2, h, N);
    k_agg_init<<<gNF, B, 0, stream>>>(h, dis, b2, agg, nf);
    k_scatter<<<gEdgeWave, B, 0, stream>>>(h, dis, ei, agg, E);
    k_relu_bf16<<<gNF, B, 0, stream>>>(agg, hbf, nf);

    // mean-pool + head
    k_pool_init<<<(G * DH + 255) / 256, B, 0, stream>>>(pooled, cnt, G);
    k_pool<<<gNF, B, 0, stream>>>(hbf, bat, pooled, cnt, nf);
    k_head<<<(G * 8 + 255) / 256, B, 0, stream>>>(pooled, cnt, Wfc, bfc, out, G);
}
